// NemotronHAttention_5360119185604
// MI455X (gfx1250) — compile-verified, matrix operand order checked by reading, not run on hardware
//
#include <hip/hip_runtime.h>
#include <hip/hip_bf16.h>

// ---------------- problem constants ----------------
#define HIDDEN    4096
#define N_HEADS   32
#define N_KVH     8
#define HEAD_DIM  128
#define N_GROUPS  4
#define Q_SIZE    4096
#define KV_SIZE   1024
#define QKV_N     6144
#define BATCH     2
#define T_SEQ     2048
#define M_ROWS    (BATCH * T_SEQ)
#define SCALING   0.08838834764831845f
#define NEG_BIG   (-3.0e38f)
#define NEG_TH    (-1.0e37f)

typedef __attribute__((ext_vector_type(16))) _Float16 v16h;
typedef __attribute__((ext_vector_type(8)))  _Float16 v8h;
typedef __attribute__((ext_vector_type(8)))  float    v8f;
typedef __attribute__((ext_vector_type(4)))  float    f32x4;

union V16H2 { v16h v; v8h q[2]; };

// Fragment load: lane's 16 halves = two contiguous 16B chunks at [0..7], [16..23].
// A-frag: lanebase = tile + (lane&15)*ld + 8*(lane>>4)        (row-major A)
// B-frag: same when element (k,n) lives at src[n*ld + k]      (K-contiguous)
__device__ __forceinline__ v16h ld_frag16(const _Float16* lanebase) {
    V16H2 f;
    f.q[0] = *(const v8h*)(lanebase);
    f.q[1] = *(const v8h*)(lanebase + 16);
    return f.v;
}

#define WMMA_F32_F16(A, B, C) \
    __builtin_amdgcn_wmma_f32_16x16x32_f16(false, (A), false, (B), (short)0, (C), false, false)

// Async copy of a 64x32-half B panel (4KB) into LDS: 256 16B chunks, 128 threads,
// exactly 2 unconditional global_load_async_to_lds_b128 per thread (no exec-mask
// guards; ASYNCcnt += 2 per wave).
__device__ __forceinline__ void async_b_panel(const _Float16* __restrict__ WT, size_t ldk,
                                              int ct0, int k0, _Float16* buf, int tid) {
    {
        const int c = tid;                 // chunks 0..127
        const int n = c >> 2, q = c & 3;
        const _Float16* g = WT + (size_t)(ct0 + n) * ldk + k0 + 8 * q;
        unsigned lds = (unsigned)(uintptr_t)(buf + n * 32 + 8 * q);
        asm volatile("global_load_async_to_lds_b128 %0, %1, off"
                     :: "v"(lds), "v"((unsigned long long)(uintptr_t)g)
                     : "memory");
    }
    {
        const int c = tid + 128;           // chunks 128..255
        const int n = c >> 2, q = c & 3;
        const _Float16* g = WT + (size_t)(ct0 + n) * ldk + k0 + 8 * q;
        unsigned lds = (unsigned)(uintptr_t)(buf + n * 32 + 8 * q);
        asm volatile("global_load_async_to_lds_b128 %0, %1, off"
                     :: "v"(lds), "v"((unsigned long long)(uintptr_t)g)
                     : "memory");
    }
}

// ============ pre-pass: hidden f32 -> f16 ============
__global__ void k_cvt_hidden(const float* __restrict__ in, _Float16* __restrict__ out) {
    long i = (long)blockIdx.x * blockDim.x + threadIdx.x;
    const long total = (long)M_ROWS * HIDDEN / 8;
    if (i >= total) return;
    f32x4 a = ((const f32x4*)in)[2 * i];
    f32x4 b = ((const f32x4*)in)[2 * i + 1];
    v8h o;
#pragma unroll
    for (int j = 0; j < 4; ++j) { o[j] = (_Float16)a[j]; o[4 + j] = (_Float16)b[j]; }
    ((v8h*)out)[i] = o;
}

// ============ pre-pass: W [K][N] f32 -> W^T [N][K] f16 ============
__global__ void k_transpose_w(const float* __restrict__ in, _Float16* __restrict__ out,
                              int K, int N) {
    __shared__ float tile[32][33];
    const int tx = threadIdx.x, ty = threadIdx.y;
    const int bk = blockIdx.y * 32, bn = blockIdx.x * 32;
#pragma unroll
    for (int i = 0; i < 4; ++i)
        tile[ty + i * 8][tx] = in[(size_t)(bk + ty + i * 8) * N + bn + tx];
    __syncthreads();
#pragma unroll
    for (int i = 0; i < 4; ++i)
        out[(size_t)(bn + ty + i * 8) * K + bk + tx] = (_Float16)tile[tx][ty + i * 8];
}

// ============ pre-pass: V slice of qkv -> vT [B][KV_SIZE][T] f16 ============
__global__ void k_vtrans(const _Float16* __restrict__ qkv, _Float16* __restrict__ vT) {
    __shared__ _Float16 tile[32][33];
    const int tx = threadIdx.x, ty = threadIdx.y;
    const int bc = blockIdx.x * 32, bt = blockIdx.y * 32, bb = blockIdx.z;
#pragma unroll
    for (int i = 0; i < 4; ++i)
        tile[ty + i * 8][tx] =
            qkv[(size_t)(bb * T_SEQ + bt + ty + i * 8) * QKV_N + Q_SIZE + KV_SIZE + bc + tx];
    __syncthreads();
#pragma unroll
    for (int i = 0; i < 4; ++i)
        vT[((size_t)bb * KV_SIZE + bc + ty + i * 8) * T_SEQ + bt + tx] = tile[tx][ty + i * 8];
}

// ============ Kernel: QKV GEMM  X16[4096,4096] @ WT[6144,4096] -> qkv f16 ============
// Block = 4 waves, 128x64 tile; B panel double-buffered in LDS via async-to-LDS.
__global__ __launch_bounds__(128) void k_gemm_qkv(const _Float16* __restrict__ X,
                                                  const _Float16* __restrict__ WT,
                                                  _Float16* __restrict__ out) {
    __shared__ __align__(16) _Float16 ldsB[2][64 * 32];     // 8 KB
    const int tid  = threadIdx.x;
    const int lane = tid & 31, wl = tid >> 5;
    const int tilesN = QKV_N / 64;                          // 96
    const int br  = blockIdx.x / tilesN;                    // 128-row block tile
    const int ct0 = (blockIdx.x % tilesN) * 64;
    const int m = lane & 15, kh = lane >> 4;

    v8f acc[2][4] = {};
    const _Float16* A0 = X + (size_t)(br * 128 + wl * 32 + m) * HIDDEN + 8 * kh;
    const _Float16* A1 = A0 + (size_t)16 * HIDDEN;

    const int nsteps = HIDDEN / 32;                         // 128
    async_b_panel(WT, HIDDEN, ct0, 0, ldsB[0], tid);
    for (int i = 0; i < nsteps; ++i) {
        const int k0 = i * 32;
        if (i + 1 < nsteps) {
            async_b_panel(WT, HIDDEN, ct0, k0 + 32, ldsB[(i + 1) & 1], tid);
            asm volatile("s_wait_asynccnt 0x2" ::: "memory");
        } else {
            asm volatile("s_wait_asynccnt 0x0" ::: "memory");
        }
        __syncthreads();                                    // panel i ready for all waves
        __builtin_prefetch(A0 + k0 + 256, 0, 1);
        v16h a0 = ld_frag16(A0 + k0);
        v16h a1 = ld_frag16(A1 + k0);
        const _Float16* Bp = ldsB[i & 1] + m * 32 + 8 * kh;
#pragma unroll
        for (int t = 0; t < 4; ++t) {
            v16h b = ld_frag16(Bp + t * 16 * 32);           // ds_load_b128 x2
            acc[0][t] = WMMA_F32_F16(a0, b, acc[0][t]);
            acc[1][t] = WMMA_F32_F16(a1, b, acc[1][t]);
        }
        __syncthreads();                                    // reads done before overwrite
    }
    const int r0 = br * 128 + wl * 32;
#pragma unroll
    for (int rt = 0; rt < 2; ++rt)
#pragma unroll
        for (int t = 0; t < 4; ++t)
#pragma unroll
            for (int v = 0; v < 8; ++v)
                out[(size_t)(r0 + rt * 16 + v + 8 * kh) * QKV_N + ct0 + t * 16 + m] =
                    (_Float16)acc[rt][t][v];
}

// ============ Kernel: RoPE on Q/K halves (in place) ============
__global__ void k_rope(_Float16* __restrict__ qkv) {
    long idx = (long)blockIdx.x * blockDim.x + threadIdx.x;
    const long total = (long)M_ROWS * (N_HEADS + N_KVH) * 64;
    if (idx >= total) return;
    int d   = idx & 63;
    int ht  = (idx >> 6) % (N_HEADS + N_KVH);
    int row = (int)(idx / (64 * (N_HEADS + N_KVH)));
    int off = (ht < N_HEADS) ? ht * HEAD_DIM : Q_SIZE + (ht - N_HEADS) * HEAD_DIM;
    int t   = row % T_SEQ;
    float ang = (float)t * __powf(10000.0f, -(float)d * (1.0f / 64.0f));
    float c, s;
    __sincosf(ang, &s, &c);
    _Float16* p = qkv + (size_t)row * QKV_N + off;
    float x1 = (float)p[d], x2 = (float)p[d + 64];
    p[d]      = (_Float16)(x1 * c - x2 * s);
    p[d + 64] = (_Float16)(x2 * c + x1 * s);
}

// ============ Kernel: causal GQA flash attention ============
__global__ __launch_bounds__(128) void k_attn(const _Float16* __restrict__ qkv,
                                              const _Float16* __restrict__ vT,
                                              _Float16* __restrict__ y) {
    __shared__ __align__(16) _Float16 s_p[4][16 * 32];
    const int lane = threadIdx.x & 31;
    const int wl   = threadIdx.x >> 5;
    const int wave = blockIdx.x * 4 + wl;
    const int qtiles = T_SEQ / 16;
    const int qt = wave % qtiles;
    const int h  = (wave / qtiles) % N_HEADS;
    const int bb = wave / (qtiles * N_HEADS);
    const int kvh = h / N_GROUPS;
    const int kh = lane >> 4, nloc = lane & 15;
    const int rowbase = bb * T_SEQ + qt * 16;

    v16h qf[4];
    {
        const _Float16* qsrc = qkv + (size_t)rowbase * QKV_N + h * HEAD_DIM
                             + (size_t)nloc * QKV_N + 8 * kh;
        const _Float16 sc = (_Float16)SCALING;
#pragma unroll
        for (int kc = 0; kc < 4; ++kc) {
            V16H2 f;
            f.q[0] = *(const v8h*)(qsrc + kc * 32)      * sc;
            f.q[1] = *(const v8h*)(qsrc + kc * 32 + 16) * sc;
            qf[kc] = f.v;
        }
    }

    float rm[8], rl[8];
    v8f o[8] = {};
#pragma unroll
    for (int v = 0; v < 8; ++v) { rm[v] = NEG_BIG; rl[v] = 0.0f; }

    const int smax = qt * 16 + 15;
    const _Float16* Kb = qkv + (size_t)bb * T_SEQ * QKV_N + Q_SIZE + kvh * HEAD_DIM
                       + (size_t)nloc * QKV_N + 8 * kh;
    const _Float16* Vt = vT + ((size_t)bb * KV_SIZE + kvh * HEAD_DIM + nloc) * T_SEQ
                       + 8 * kh;
    _Float16* L = s_p[wl];

    for (int s0 = 0; s0 <= smax; s0 += 32) {
        v8f sc2[2] = {};
#pragma unroll
        for (int st = 0; st < 2; ++st) {
            const _Float16* kb = Kb + (size_t)(s0 + st * 16) * QKV_N;
#pragma unroll
            for (int kc = 0; kc < 4; ++kc) {
                v16h b = ld_frag16(kb + kc * 32);
                sc2[st] = WMMA_F32_F16(qf[kc], b, sc2[st]);
            }
        }
#pragma unroll
        for (int v = 0; v < 8; ++v) {
            int row = qt * 16 + v + 8 * kh;
            if (s0 + nloc      > row) sc2[0][v] = NEG_BIG;
            if (s0 + 16 + nloc > row) sc2[1][v] = NEG_BIG;
            float mx = fmaxf(sc2[0][v], sc2[1][v]);
#pragma unroll
            for (int off = 1; off < 16; off <<= 1)
                mx = fmaxf(mx, __shfl_xor(mx, off, 16));
            float mnew = fmaxf(rm[v], mx);
            float fs = (rm[v] <= NEG_TH) ? 0.0f : __expf(rm[v] - mnew);
            float p0 = (sc2[0][v] <= NEG_TH) ? 0.0f : __expf(sc2[0][v] - mnew);
            float p1 = (sc2[1][v] <= NEG_TH) ? 0.0f : __expf(sc2[1][v] - mnew);
            float sum = p0 + p1;
#pragma unroll
            for (int off = 1; off < 16; off <<= 1)
                sum += __shfl_xor(sum, off, 16);
            rl[v] = rl[v] * fs + sum;
            rm[v] = mnew;
#pragma unroll
            for (int ct = 0; ct < 8; ++ct) o[ct][v] *= fs;
            L[(v + 8 * kh) * 32 + nloc]      = (_Float16)p0;
            L[(v + 8 * kh) * 32 + 16 + nloc] = (_Float16)p1;
        }
        asm volatile("s_wait_dscnt 0x0" ::: "memory");
        v16h pf = ld_frag16(L + nloc * 32 + 8 * kh);
#pragma unroll
        for (int ct = 0; ct < 8; ++ct) {
            v16h vb = ld_frag16(Vt + (size_t)ct * 16 * T_SEQ + s0);
            o[ct] = WMMA_F32_F16(pf, vb, o[ct]);
        }
    }

#pragma unroll
    for (int v = 0; v < 8; ++v) {
        float inv = (rl[v] > 0.0f) ? (1.0f / rl[v]) : 0.0f;
        int mr = v + 8 * kh;
#pragma unroll
        for (int ct = 0; ct < 8; ++ct)
            y[(size_t)(rowbase + mr) * Q_SIZE + h * HEAD_DIM + ct * 16 + nloc] =
                (_Float16)(o[ct][v] * inv);
    }
}

// ============ Kernel: output GEMM  y[4096,4096]f16 @ WoT[4096,4096]f16 -> out f32 ============
__global__ __launch_bounds__(128) void k_gemm_o(const _Float16* __restrict__ Y,
                                                const _Float16* __restrict__ WT,
                                                float* __restrict__ out) {
    __shared__ __align__(16) _Float16 ldsB[2][64 * 32];
    const int tid  = threadIdx.x;
    const int lane = tid & 31, wl = tid >> 5;
    const int tilesN = HIDDEN / 64;                         // 64
    const int br  = blockIdx.x / tilesN;
    const int ct0 = (blockIdx.x % tilesN) * 64;
    const int m = lane & 15, kh = lane >> 4;

    v8f acc[2][4] = {};
    const _Float16* A0 = Y + (size_t)(br * 128 + wl * 32 + m) * Q_SIZE + 8 * kh;
    const _Float16* A1 = A0 + (size_t)16 * Q_SIZE;

    const int nsteps = Q_SIZE / 32;
    async_b_panel(WT, Q_SIZE, ct0, 0, ldsB[0], tid);
    for (int i = 0; i < nsteps; ++i) {
        const int k0 = i * 32;
        if (i + 1 < nsteps) {
            async_b_panel(WT, Q_SIZE, ct0, k0 + 32, ldsB[(i + 1) & 1], tid);
            asm volatile("s_wait_asynccnt 0x2" ::: "memory");
        } else {
            asm volatile("s_wait_asynccnt 0x0" ::: "memory");
        }
        __syncthreads();
        __builtin_prefetch(A0 + k0 + 256, 0, 1);
        v16h a0 = ld_frag16(A0 + k0);
        v16h a1 = ld_frag16(A1 + k0);
        const _Float16* Bp = ldsB[i & 1] + m * 32 + 8 * kh;
#pragma unroll
        for (int t = 0; t < 4; ++t) {
            v16h b = ld_frag16(Bp + t * 16 * 32);
            acc[0][t] = WMMA_F32_F16(a0, b, acc[0][t]);
            acc[1][t] = WMMA_F32_F16(a1, b, acc[1][t]);
        }
        __syncthreads();
    }
    const int r0 = br * 128 + wl * 32;
#pragma unroll
    for (int rt = 0; rt < 2; ++rt)
#pragma unroll
        for (int t = 0; t < 4; ++t)
#pragma unroll
            for (int v = 0; v < 8; ++v)
                out[(size_t)(r0 + rt * 16 + v + 8 * kh) * HIDDEN + ct0 + t * 16 + m] =
                    acc[rt][t][v];
}

// ---------------- host launcher ----------------
extern "C" void kernel_launch(void* const* d_in, const int* in_sizes, int n_in,
                              void* d_out, int out_size, void* d_ws, size_t ws_size,
                              hipStream_t stream) {
    const float* hidden = (const float*)d_in[0];
    const float* w_qkv  = (const float*)d_in[1];
    const float* w_o    = (const float*)d_in[2];
    float*       out    = (float*)d_out;

    _Float16* qkv   = (_Float16*)d_ws;                                // 4096*6144
    _Float16* y     = qkv   + (size_t)M_ROWS * QKV_N;                 // 4096*4096
    _Float16* x16   = y     + (size_t)M_ROWS * Q_SIZE;                // 4096*4096
    _Float16* wqkvT = x16   + (size_t)M_ROWS * HIDDEN;                // 6144*4096
    _Float16* woT   = wqkvT + (size_t)QKV_N * HIDDEN;                 // 4096*4096
    _Float16* vT    = woT   + (size_t)HIDDEN * Q_SIZE;                // 2*1024*2048

    {
        long total = (long)M_ROWS * HIDDEN / 8;
        k_cvt_hidden<<<(int)((total + 255) / 256), 256, 0, stream>>>(hidden, x16);
    }
    k_transpose_w<<<dim3(QKV_N / 32, HIDDEN / 32), dim3(32, 8), 0, stream>>>(w_qkv, wqkvT, HIDDEN, QKV_N);
    k_transpose_w<<<dim3(HIDDEN / 32, Q_SIZE / 32), dim3(32, 8), 0, stream>>>(w_o, woT, Q_SIZE, HIDDEN);

    // 1) QKV projection: (4096/128)*(6144/64) = 3072 blocks of 4 waves
    k_gemm_qkv<<<(M_ROWS / 128) * (QKV_N / 64), 128, 0, stream>>>(x16, wqkvT, qkv);

    // 2) RoPE
    {
        long total = (long)M_ROWS * (N_HEADS + N_KVH) * 64;
        k_rope<<<(int)((total + 255) / 256), 256, 0, stream>>>(qkv);
    }

    // 3) V transpose
    k_vtrans<<<dim3(KV_SIZE / 32, T_SEQ / 32, BATCH), dim3(32, 8), 0, stream>>>(qkv, vT);

    // 4) flash attention
    k_attn<<<BATCH * N_HEADS * (T_SEQ / 16) / 4, 128, 0, stream>>>(qkv, vT, y);

    // 5) output projection: (4096/128)*(4096/64) = 2048 blocks
    k_gemm_o<<<(M_ROWS / 128) * (HIDDEN / 64), 128, 0, stream>>>(y, woT, out);
}
